// TensorBase_489626272093
// MI455X (gfx1250) — matrix operand order repaired and checked by reading, not compiled, not run on hardware
//
#include <hip/hip_runtime.h>
#include <hip/hip_bf16.h>

typedef __attribute__((ext_vector_type(16))) _Float16 v16h;
typedef __attribute__((ext_vector_type(8)))  float    v8f;
typedef __attribute__((ext_vector_type(4)))  float    f32x4;

#define GRID 128
#define GRID2 (128 * 128)
#define GRID3 (128 * 128 * 128)
#define NRAY 4096
#define NSAMP 256
#define KPAD 192   // 174 padded to 6 k-steps of 32
#define FC 128

// ---- LDS layout (bytes) ----
#define OFF_W1T   0            // 128 x 192 f16 = 49152
#define OFF_W2T   49152        // 128 x 128 f16 = 32768
#define OFF_W3T   81920        // 16 x 128 f16  = 4096
#define OFF_FEAT  86016        // 256 x 192 f16 = 98304
#define OFF_H     184320       // 16 waves * 16*128 f16 = 65536
#define OFF_SIG   249856       // 256 f32 = 1024
#define OFF_WGT   250880       // 256 f32 = 1024
#define OFF_ACC   251904       // 3 f32 (+pad) = 16
#define SMEM_BYTES 251920

#define WQ_TOTAL (86016 / 16)  // f32x4 count of weight staging region

// ---- weight prep: transpose + f32->f16, pad K to 192 and W3 cols to 16 ----
__global__ void prep_weights(const float* __restrict__ W1,
                             const float* __restrict__ W2,
                             const float* __restrict__ W3,
                             _Float16* __restrict__ wsHalf) {
  int i = blockIdx.x * 256 + threadIdx.x;
  if (i < 24576) {                       // W1t [n=128][k=192]
    int n = i / KPAD, k = i % KPAD;
    wsHalf[i] = (_Float16)((k < 174) ? W1[k * FC + n] : 0.0f);
  } else if (i < 40960) {                // W2t [n=128][k=128]
    int j = i - 24576;
    int n = j / FC, k = j % FC;
    wsHalf[i] = (_Float16)W2[k * FC + n];
  } else if (i < 43008) {                // W3t [n=16][k=128]
    int j = i - 40960;
    int n = j / FC, k = j % FC;
    wsHalf[i] = (_Float16)((n < 3) ? W3[k * 3 + n] : 0.0f);
  }
}

__device__ __forceinline__ float softplus_shift(float x) {
  return (x > 20.0f) ? x : log1pf(expf(x));
}

// one block = one ray (256 samples), 512 threads = 16 waves
__global__ __launch_bounds__(512)
void render_kernel(const float* __restrict__ xyz,
                   const float* __restrict__ viewdirs,
                   const float* __restrict__ dists,
                   const float* __restrict__ alphaV,
                   const float* __restrict__ densV,
                   const float* __restrict__ appV,
                   const float* __restrict__ b1,
                   const float* __restrict__ b2,
                   const float* __restrict__ b3,
                   const _Float16* __restrict__ wsHalf,
                   float* __restrict__ out) {
  extern __shared__ char smem[];
  _Float16* sW1   = (_Float16*)(smem + OFF_W1T);
  _Float16* sW2   = (_Float16*)(smem + OFF_W2T);
  _Float16* sW3   = (_Float16*)(smem + OFF_W3T);
  _Float16* sFeat = (_Float16*)(smem + OFF_FEAT);
  _Float16* sH    = (_Float16*)(smem + OFF_H);
  float*    sSig  = (float*)(smem + OFF_SIG);
  float*    sWgt  = (float*)(smem + OFF_WGT);
  float*    sAcc  = (float*)(smem + OFF_ACC);

  const int ray = blockIdx.x;
  const int t   = threadIdx.x;

  if (t < 3) sAcc[t] = 0.0f;

  if (t >= 256) {
    // ---- cooperative weight staging: global (prepped) -> LDS ----
    f32x4* dst = (f32x4*)smem;
    const f32x4* src = (const f32x4*)wsHalf;
    for (int i = t - 256; i < WQ_TOTAL; i += 256) dst[i] = src[i];
  } else {
    // ---- per-sample trilerp + feature build ----
    const int s = t;
    const float* pp = xyz + ((size_t)ray * NSAMP + s) * 3;
    float px = pp[0], py = pp[1], pz = pp[2];
    float gx = (px + 1.0f) * 0.5f * (GRID - 1);
    float gy = (py + 1.0f) * 0.5f * (GRID - 1);
    float gz = (pz + 1.0f) * 0.5f * (GRID - 1);
    int x0 = min(max((int)floorf(gx), 0), GRID - 2);
    int y0 = min(max((int)floorf(gy), 0), GRID - 2);
    int z0 = min(max((int)floorf(gz), 0), GRID - 2);
    float wx = gx - (float)x0, wy = gy - (float)y0, wz = gz - (float)z0;
    float ox = 1.0f - wx, oy = 1.0f - wy, oz = 1.0f - wz;

    int off[8];
    off[0] = z0 * GRID2 + y0 * GRID + x0;
    off[1] = off[0] + 1;
    off[2] = off[0] + GRID;
    off[3] = off[2] + 1;
    off[4] = off[0] + GRID2;
    off[5] = off[4] + 1;
    off[6] = off[4] + GRID;
    off[7] = off[6] + 1;
    float wt[8];
    wt[0] = oz * oy * ox; wt[1] = oz * oy * wx;
    wt[2] = oz * wy * ox; wt[3] = oz * wy * wx;
    wt[4] = wz * oy * ox; wt[5] = wz * oy * wx;
    wt[6] = wz * wy * ox; wt[7] = wz * wy * wx;

    float av = 0.0f, dv = 0.0f;
#pragma unroll
    for (int k = 0; k < 8; k++) {
      av += alphaV[off[k]] * wt[k];
      dv += densV[off[k]] * wt[k];
    }
    float sig = softplus_shift(dv - 10.0f);
    sSig[s] = (av > 0.001f) ? sig : 0.0f;

    float af[27];
#pragma unroll 1
    for (int c = 0; c < 27; c++) {
      const float* v = appV + (size_t)c * GRID3;
      float acc = 0.0f;
#pragma unroll
      for (int k = 0; k < 8; k++) acc += v[off[k]] * wt[k];
      af[c] = acc;
    }
    const float* vdp = viewdirs + (size_t)ray * 3;
    float vd[3] = {vdp[0], vdp[1], vdp[2]};

    // feature row: [app 27 | vd 3 | sin(app pe2) 54 | cos 54 | sin(vd pe6) 18 | cos 18 | pad]
    _Float16* fr = sFeat + s * KPAD;
#pragma unroll
    for (int c = 0; c < 27; c++) fr[c] = (_Float16)af[c];
    fr[27] = (_Float16)vd[0]; fr[28] = (_Float16)vd[1]; fr[29] = (_Float16)vd[2];
#pragma unroll
    for (int c = 0; c < 27; c++) {
#pragma unroll
      for (int f = 0; f < 2; f++) {
        float pv = af[c] * (float)(1 << f);
        fr[30 + c * 2 + f] = (_Float16)sinf(pv);
        fr[84 + c * 2 + f] = (_Float16)cosf(pv);
      }
    }
#pragma unroll
    for (int c = 0; c < 3; c++) {
#pragma unroll
      for (int f = 0; f < 6; f++) {
        float pv = vd[c] * (float)(1 << f);
        fr[138 + c * 6 + f] = (_Float16)sinf(pv);
        fr[156 + c * 6 + f] = (_Float16)cosf(pv);
      }
    }
#pragma unroll
    for (int j = 174; j < KPAD; j++) fr[j] = (_Float16)0.0f;
  }
  __syncthreads();

  // ---- transmittance scan (sequential, 256 steps) ----
  if (t == 0) {
    const float* dr = dists + (size_t)ray * NSAMP;
    float T = 1.0f;
    for (int s = 0; s < NSAMP; s++) {
      float a = 1.0f - expf(-sSig[s] * dr[s] * 25.0f);
      sWgt[s] = a * T;
      T *= (1.0f - a + 1e-10f);
    }
  }
  __syncthreads();

  // ---- MLP via WMMA: each wave owns a 16-row M-tile ----
  const int wid  = t >> 5;
  const int lane = t & 31;
  const int lo   = lane & 15;
  const int hi   = lane >> 4;
  const int mbase = wid * 16;
  _Float16* hRow = sH + wid * (16 * FC);

  union FragU { v16h v; f32x4 q[2]; };

  // layer 1: [16x192] x [192x128]
  v16h a1[6];
  {
    const _Float16* fr = sFeat + (mbase + lo) * KPAD;
#pragma unroll
    for (int ks = 0; ks < 6; ks++) {
      int c0 = ks * 32 + 8 * hi;
      FragU u;
      u.q[0] = *(const f32x4*)(fr + c0);
      u.q[1] = *(const f32x4*)(fr + c0 + 16);
      a1[ks] = u.v;
    }
  }
#pragma unroll
  for (int n = 0; n < 8; n++) {
    v8f c = {};
    const _Float16* wr = sW1 + (n * 16 + lo) * KPAD;
#pragma unroll
    for (int ks = 0; ks < 6; ks++) {
      FragU u;
      const _Float16* bp = wr + ks * 32 + 8 * hi;
      u.q[0] = *(const f32x4*)(bp);
      u.q[1] = *(const f32x4*)(bp + 16);
      c = __builtin_amdgcn_wmma_f32_16x16x32_f16(false, a1[ks], false, u.v,
                                                 (short)0, c, false, false);
    }
    float bias = b1[n * 16 + lo];
#pragma unroll
    for (int r = 0; r < 8; r++) {
      float v = c[r] + bias;
      v = v > 0.0f ? v : 0.0f;
      hRow[(hi * 8 + r) * FC + n * 16 + lo] = (_Float16)v;
    }
  }

  // layer 2: [16x128] x [128x128]
  v16h a2[4];
  {
    const _Float16* hr = hRow + lo * FC;
#pragma unroll
    for (int ks = 0; ks < 4; ks++) {
      int c0 = ks * 32 + 8 * hi;
      FragU u;
      u.q[0] = *(const f32x4*)(hr + c0);
      u.q[1] = *(const f32x4*)(hr + c0 + 16);
      a2[ks] = u.v;
    }
  }
#pragma unroll
  for (int n = 0; n < 8; n++) {
    v8f c = {};
    const _Float16* wr = sW2 + (n * 16 + lo) * FC;
#pragma unroll
    for (int ks = 0; ks < 4; ks++) {
      FragU u;
      const _Float16* bp = wr + ks * 32 + 8 * hi;
      u.q[0] = *(const f32x4*)(bp);
      u.q[1] = *(const f32x4*)(bp + 16);
      c = __builtin_amdgcn_wmma_f32_16x16x32_f16(false, a2[ks], false, u.v,
                                                 (short)0, c, false, false);
    }
    float bias = b2[n * 16 + lo];
#pragma unroll
    for (int r = 0; r < 8; r++) {
      float v = c[r] + bias;
      v = v > 0.0f ? v : 0.0f;
      hRow[(hi * 8 + r) * FC + n * 16 + lo] = (_Float16)v;
    }
  }

  // layer 3: [16x128] x [128x16] (cols 0..2 valid)
  v16h a3[4];
  {
    const _Float16* hr = hRow + lo * FC;
#pragma unroll
    for (int ks = 0; ks < 4; ks++) {
      int c0 = ks * 32 + 8 * hi;
      FragU u;
      u.q[0] = *(const f32x4*)(hr + c0);
      u.q[1] = *(const f32x4*)(hr + c0 + 16);
      a3[ks] = u.v;
    }
  }
  v8f c3 = {};
  {
    const _Float16* wr = sW3 + lo * FC;
#pragma unroll
    for (int ks = 0; ks < 4; ks++) {
      FragU u;
      const _Float16* bp = wr + ks * 32 + 8 * hi;
      u.q[0] = *(const f32x4*)(bp);
      u.q[1] = *(const f32x4*)(bp + 16);
      c3 = __builtin_amdgcn_wmma_f32_16x16x32_f16(false, a3[ks], false, u.v,
                                                  (short)0, c3, false, false);
    }
  }

  // sigmoid + weighted accumulation into per-block rgb
  if (lo < 3) {
    float bias = b3[lo];
    float partial = 0.0f;
#pragma unroll
    for (int r = 0; r < 8; r++) {
      int M = hi * 8 + r;
      float z = c3[r] + bias;
      float rgb = 1.0f / (1.0f + expf(-z));
      partial += sWgt[mbase + M] * rgb;
    }
    atomicAdd(&sAcc[lo], partial);
  }
  __syncthreads();
  if (t < 3) out[(size_t)ray * 3 + t] = sAcc[t];
}

extern "C" void kernel_launch(void* const* d_in, const int* in_sizes, int n_in,
                              void* d_out, int out_size, void* d_ws, size_t ws_size,
                              hipStream_t stream) {
  const float* xyz      = (const float*)d_in[0];
  const float* viewdirs = (const float*)d_in[1];
  const float* dists    = (const float*)d_in[2];
  const float* alphaV   = (const float*)d_in[3];
  const float* densV    = (const float*)d_in[4];
  const float* appV     = (const float*)d_in[5];
  const float* W1       = (const float*)d_in[6];
  const float* b1       = (const float*)d_in[7];
  const float* W2       = (const float*)d_in[8];
  const float* b2       = (const float*)d_in[9];
  const float* W3       = (const float*)d_in[10];
  const float* b3       = (const float*)d_in[11];
  _Float16* wsHalf = (_Float16*)d_ws;
  float* out = (float*)d_out;

  prep_weights<<<43008 / 256, 256, 0, stream>>>(W1, W2, W3, wsHalf);

  (void)hipFuncSetAttribute((const void*)render_kernel,
                            hipFuncAttributeMaxDynamicSharedMemorySize,
                            SMEM_BYTES);
  render_kernel<<<NRAY, 512, SMEM_BYTES, stream>>>(
      xyz, viewdirs, dists, alphaV, densV, appV, b1, b2, b3, wsHalf, out);
}
